// MatchingLayer_2353642078946
// MI455X (gfx1250) — matrix operand order repaired and checked by reading, not compiled, not run on hardware
//
#include <hip/hip_runtime.h>

// ---------------------------------------------------------------------------
// MatchingLayer for MI455X (gfx1250, wave32, WMMA bf16)
//   out = LayerNorm( [X, softmax(mask(X Xᵀ/√H)) X] · W + b ) * gamma + beta
// Workspace layout (assumes ws_size >= 100 MB):
//   [0,       32MB)  Xr  : bf16 X row-major        [B][S][H]
//   [32MB,    64MB)  Xt  : bf16 X transposed       [B][H][S]
//   [64MB,    68MB)  WbT : bf16 W transposed       [H][2H]
//   [68MB,   100MB)  Ctx : bf16 attention context  [B][S][H]
// Tiling: 256-thread WGs (8 waves), M=32 rows (2x16 M-subtiles per wave) so
// each B fragment feeds 2 WMMAs; fragment loads are software-pipelined.
// ---------------------------------------------------------------------------

typedef __attribute__((ext_vector_type(16))) __bf16 v16bf;
typedef __attribute__((ext_vector_type(8)))  float  v8f;

#define LOG2E 1.4426950408889634f

static constexpr int B_ = 8;
static constexpr int S_ = 2048;
static constexpr int H_ = 1024;

__device__ __forceinline__ v16bf load_frag(const void* p0, const void* p1) {
  union U { int4 q[2]; v16bf v; } u;
  u.q[0] = *(const int4*)p0;
  u.q[1] = *(const int4*)p1;
  return u.v;
}

// ------------------------------ fp32 -> bf16 -------------------------------
__global__ void k_convert(const float* __restrict__ s, __bf16* __restrict__ d, int n) {
  int i = blockIdx.x * blockDim.x + threadIdx.x;
  if (i < n) d[i] = (__bf16)s[i];
}

// ------------------- tiled transpose fp32 -> bf16 (dst[c][r]) --------------
__global__ void k_transpose(const float* __restrict__ src, __bf16* __restrict__ dst,
                            int rows, int cols, size_t bsrc, size_t bdst) {
  __shared__ float t[32][33];
  const int b  = blockIdx.z;
  const int c0 = blockIdx.x * 32;
  const int r0 = blockIdx.y * 32;
  const float* s = src + (size_t)b * bsrc;
  __bf16*      d = dst + (size_t)b * bdst;
#pragma unroll
  for (int j = 0; j < 4; ++j) {
    int r = r0 + threadIdx.y + j * 8;
    t[threadIdx.y + j * 8][threadIdx.x] = s[(size_t)r * cols + c0 + threadIdx.x];
  }
  __syncthreads();
#pragma unroll
  for (int j = 0; j < 4; ++j) {
    int c = c0 + threadIdx.y + j * 8;
    d[(size_t)c * rows + r0 + threadIdx.x] = (__bf16)t[threadIdx.x][threadIdx.y + j * 8];
  }
}

// ----------------------- flash attention -> context ------------------------
// Grid: (S/32, B). Block: 256 threads = 8 waves. WG: 32 query rows.
// Per 128-key iteration, wave w computes S-tiles for keys [w*16, w*16+16)
// against both 16-row M-subtiles, and owns O columns [w*128, w*128+128).
__global__ void __launch_bounds__(256)
k_attn(const __bf16* __restrict__ Xr, const __bf16* __restrict__ Xt,
       const float* __restrict__ masks, __bf16* __restrict__ Ctx) {
  const int b    = blockIdx.y;
  const int q0   = blockIdx.x * 32;
  const int tid  = threadIdx.x;
  const int wave = tid >> 5;
  const int lane = tid & 31;
  const int ln   = lane & 15;
  const int hi   = lane >> 4;

  __shared__ __align__(16) unsigned char qlds[32 * 2064]; // 32 rows x 1024 bf16 (+16B pad)
  __shared__ __align__(16) unsigned char plds[32 * 272];  // 32 rows x 128 bf16 (+16B pad)
  __shared__ float wmax[8][32];
  __shared__ float wsum[8][32];
  __shared__ float mrun[32];
  __shared__ float lrun[32];

  // stage Q tile (32 x 1024 bf16) into LDS, padded rows
  for (int c = tid; c < 4096; c += 256) {
    int row = c >> 7;
    int off = (c & 127) * 16;
    *(int4*)(qlds + row * 2064 + off) =
        *(const int4*)((const char*)(Xr + ((size_t)b * S_ + q0 + row) * H_) + off);
  }
  if (tid < 32) { mrun[tid] = -1e30f; lrun[tid] = 0.f; }
  __syncthreads();

  v8f acc[2][8];
#pragma unroll
  for (int m = 0; m < 2; ++m)
#pragma unroll
    for (int t = 0; t < 8; ++t)
#pragma unroll
      for (int r = 0; r < 8; ++r) acc[m][t][r] = 0.f;

  const float scale = 0.03125f; // 1/sqrt(1024)

  for (int kb = 0; kb < S_; kb += 128) { // 16 iterations
    const int mykey = kb + wave * 16 + ln;
    const char* krow = (const char*)(Xr + ((size_t)b * S_ + mykey) * H_);

    // ---- S-tiles (2 M-subtiles x 16 keys), K=H, pipelined fragment loads ----
    v8f sacc[2];
#pragma unroll
    for (int m = 0; m < 2; ++m)
#pragma unroll
      for (int r = 0; r < 8; ++r) sacc[m][r] = 0.f;

    v16bf fa0 = load_frag(qlds + ln * 2064 + hi * 16,
                          qlds + ln * 2064 + hi * 16 + 32);
    v16bf fa1 = load_frag(qlds + (16 + ln) * 2064 + hi * 16,
                          qlds + (16 + ln) * 2064 + hi * 16 + 32);
    v16bf fb  = load_frag(krow + hi * 16, krow + hi * 16 + 32);
#pragma unroll 4
    for (int kc = 0; kc < 32; ++kc) {
      v16bf fa0n = fa0, fa1n = fa1, fbn = fb;
      if (kc < 31) {
        const unsigned char* a0 = qlds + ln * 2064 + (kc + 1) * 64 + hi * 16;
        const unsigned char* a1 = qlds + (16 + ln) * 2064 + (kc + 1) * 64 + hi * 16;
        const char*          bp = krow + (kc + 1) * 64 + hi * 16;
        fa0n = load_frag(a0, a0 + 32);
        fa1n = load_frag(a1, a1 + 32);
        fbn  = load_frag(bp, bp + 32);
      }
      sacc[0] = __builtin_amdgcn_wmma_f32_16x16x32_bf16(false, fa0, false, fb,
                                                        (short)0, sacc[0], false, false);
      sacc[1] = __builtin_amdgcn_wmma_f32_16x16x32_bf16(false, fa1, false, fb,
                                                        (short)0, sacc[1], false, false);
      fa0 = fa0n; fa1 = fa1n; fb = fbn;
    }

    // ---- scale + logit mask ----
    float sv[2][8];
#pragma unroll
    for (int m = 0; m < 2; ++m)
#pragma unroll
      for (int r = 0; r < 8; ++r) {
        int row = m * 16 + hi * 8 + r;
        float mv = masks[((size_t)b * S_ + q0 + row) * S_ + mykey];
        sv[m][r] = sacc[m][r] * scale * mv - 1e9f * (1.f - mv);
      }

    // ---- per-wave row max ----
    float mx[2][8];
#pragma unroll
    for (int m = 0; m < 2; ++m)
#pragma unroll
      for (int r = 0; r < 8; ++r) mx[m][r] = sv[m][r];
#pragma unroll
    for (int d = 8; d >= 1; d >>= 1)
#pragma unroll
      for (int m = 0; m < 2; ++m)
#pragma unroll
        for (int r = 0; r < 8; ++r)
          mx[m][r] = fmaxf(mx[m][r], __shfl_xor(mx[m][r], d, 32));
    if (ln == 0)
#pragma unroll
      for (int m = 0; m < 2; ++m)
#pragma unroll
        for (int r = 0; r < 8; ++r) wmax[wave][m * 16 + hi * 8 + r] = mx[m][r];
    __syncthreads(); // B1

    // ---- online softmax ----
    float a_[2][8], mn_[2][8], p_[2][8];
#pragma unroll
    for (int m = 0; m < 2; ++m)
#pragma unroll
      for (int r = 0; r < 8; ++r) {
        int row = m * 16 + hi * 8 + r;
        float mold = mrun[row];
        float tmax = wmax[0][row];
#pragma unroll
        for (int w = 1; w < 8; ++w) tmax = fmaxf(tmax, wmax[w][row]);
        float mn = fmaxf(mold, tmax);
        mn_[m][r] = mn;
        a_[m][r]  = __builtin_amdgcn_exp2f((mold - mn) * LOG2E);
        p_[m][r]  = __builtin_amdgcn_exp2f((sv[m][r] - mn) * LOG2E);
      }
    float rs[2][8];
#pragma unroll
    for (int m = 0; m < 2; ++m)
#pragma unroll
      for (int r = 0; r < 8; ++r) rs[m][r] = p_[m][r];
#pragma unroll
    for (int d = 8; d >= 1; d >>= 1)
#pragma unroll
      for (int m = 0; m < 2; ++m)
#pragma unroll
        for (int r = 0; r < 8; ++r) rs[m][r] += __shfl_xor(rs[m][r], d, 32);
    if (ln == 0)
#pragma unroll
      for (int m = 0; m < 2; ++m)
#pragma unroll
        for (int r = 0; r < 8; ++r) wsum[wave][m * 16 + hi * 8 + r] = rs[m][r];
#pragma unroll
    for (int m = 0; m < 2; ++m)
#pragma unroll
      for (int r = 0; r < 8; ++r)
        *((__bf16*)(plds + (m * 16 + hi * 8 + r) * 272 + (wave * 16 + ln) * 2)) =
            (__bf16)p_[m][r];
    __syncthreads(); // B2

    if (wave == 0 && ln == 0) {
#pragma unroll
      for (int m = 0; m < 2; ++m)
#pragma unroll
        for (int r = 0; r < 8; ++r) {
          int row = m * 16 + hi * 8 + r;
          float s = wsum[0][row];
#pragma unroll
          for (int w = 1; w < 8; ++w) s += wsum[w][row];
          lrun[row] = lrun[row] * a_[m][r] + s;
          mrun[row] = mn_[m][r];
        }
    }

    // ---- rescale accumulators, then O += P·V (128 cols, pipelined) ----
#pragma unroll
    for (int m = 0; m < 2; ++m)
#pragma unroll
      for (int t = 0; t < 8; ++t)
#pragma unroll
        for (int r = 0; r < 8; ++r) acc[m][t][r] *= a_[m][r];

#pragma unroll
    for (int kk = 0; kk < 4; ++kk) {
      const unsigned char* ap0 = plds + ln * 272 + kk * 64 + hi * 16;
      const unsigned char* ap1 = plds + (16 + ln) * 272 + kk * 64 + hi * 16;
      v16bf fp0 = load_frag(ap0, ap0 + 32);
      v16bf fp1 = load_frag(ap1, ap1 + 32);
      const char* v0 =
          (const char*)(Xt + ((size_t)b * H_ + wave * 128 + ln) * S_ + kb + kk * 32 + hi * 8);
      v16bf fv = load_frag(v0, v0 + 32);
#pragma unroll
      for (int t = 0; t < 8; ++t) {
        v16bf fvn = fv;
        if (t < 7) {
          const char* vp = v0 + (size_t)(t + 1) * 16 * S_ * 2;
          fvn = load_frag(vp, vp + 32);
        }
        acc[0][t] = __builtin_amdgcn_wmma_f32_16x16x32_bf16(false, fp0, false, fv,
                                                            (short)0, acc[0][t], false, false);
        acc[1][t] = __builtin_amdgcn_wmma_f32_16x16x32_bf16(false, fp1, false, fv,
                                                            (short)0, acc[1][t], false, false);
        fv = fvn;
      }
    }
  }

  __syncthreads();
  float inv[2][8];
#pragma unroll
  for (int m = 0; m < 2; ++m)
#pragma unroll
    for (int r = 0; r < 8; ++r) inv[m][r] = 1.f / lrun[m * 16 + hi * 8 + r];
#pragma unroll
  for (int m = 0; m < 2; ++m)
#pragma unroll
    for (int t = 0; t < 8; ++t) {
      int nh = wave * 128 + t * 16 + ln;
#pragma unroll
      for (int r = 0; r < 8; ++r)
        Ctx[((size_t)b * S_ + q0 + m * 16 + hi * 8 + r) * H_ + nh] =
            (__bf16)(acc[m][t][r] * inv[m][r]);
    }
}

// -------------- matching GEMM (K=2H concat split) + LayerNorm --------------
// Grid: (S/32, B). Block: 256 threads = 8 waves. Wave w owns cols [w*128, +128),
// both 16-row M-subtiles. Cross-wave LayerNorm over the full 1024 columns.
__global__ void __launch_bounds__(256)
k_match(const __bf16* __restrict__ Xr, const __bf16* __restrict__ Ctx,
        const __bf16* __restrict__ WbT, const float* __restrict__ bias,
        const float* __restrict__ gamma, const float* __restrict__ beta,
        float* __restrict__ out) {
  const int b    = blockIdx.y;
  const int q0   = blockIdx.x * 32;
  const int tid  = threadIdx.x;
  const int wave = tid >> 5;
  const int lane = tid & 31;
  const int ln   = lane & 15;
  const int hi   = lane >> 4;

  __shared__ float ssum[8][32];
  __shared__ float ssq[8][32];

  v8f acc[2][8];
#pragma unroll
  for (int m = 0; m < 2; ++m)
#pragma unroll
    for (int t = 0; t < 8; ++t)
#pragma unroll
      for (int r = 0; r < 8; ++r) acc[m][t][r] = 0.f;

  const char* arowX[2] = {
      (const char*)(Xr + ((size_t)b * S_ + q0 + ln) * H_),
      (const char*)(Xr + ((size_t)b * S_ + q0 + 16 + ln) * H_)};
  const char* arowC[2] = {
      (const char*)(Ctx + ((size_t)b * S_ + q0 + ln) * H_),
      (const char*)(Ctx + ((size_t)b * S_ + q0 + 16 + ln) * H_)};

  v16bf fa0 = load_frag(arowX[0] + hi * 16, arowX[0] + hi * 16 + 32);
  v16bf fa1 = load_frag(arowX[1] + hi * 16, arowX[1] + hi * 16 + 32);

  for (int kc = 0; kc < 64; ++kc) { // K = 2048 = concat(X, context)
    v16bf fa0n = fa0, fa1n = fa1;
    if (kc < 63) {
      int kn = kc + 1;
      const char* a0 = ((kn < 32) ? arowX[0] + kn * 64 : arowC[0] + (kn - 32) * 64) + hi * 16;
      const char* a1 = ((kn < 32) ? arowX[1] + kn * 64 : arowC[1] + (kn - 32) * 64) + hi * 16;
      fa0n = load_frag(a0, a0 + 32);
      fa1n = load_frag(a1, a1 + 32);
    }
    const char* b0 = (const char*)(WbT + (size_t)(wave * 128 + ln) * (2 * H_) +
                                   kc * 32 + hi * 8);
    v16bf fb = load_frag(b0, b0 + 32);
#pragma unroll
    for (int t = 0; t < 8; ++t) {
      v16bf fbn = fb;
      if (t < 7) {
        const char* bp = b0 + (size_t)(t + 1) * 16 * (2 * H_) * 2;
        fbn = load_frag(bp, bp + 32);
      }
      acc[0][t] = __builtin_amdgcn_wmma_f32_16x16x32_bf16(false, fa0, false, fb,
                                                          (short)0, acc[0][t], false, false);
      acc[1][t] = __builtin_amdgcn_wmma_f32_16x16x32_bf16(false, fa1, false, fb,
                                                          (short)0, acc[1][t], false, false);
      fb = fbn;
    }
    fa0 = fa0n; fa1 = fa1n;
  }

  // ---- bias + row statistics (Σx, Σx²) ----
  float sx[2][8], sq2[2][8];
#pragma unroll
  for (int m = 0; m < 2; ++m)
#pragma unroll
    for (int r = 0; r < 8; ++r) { sx[m][r] = 0.f; sq2[m][r] = 0.f; }
#pragma unroll
  for (int t = 0; t < 8; ++t) {
    int n = wave * 128 + t * 16 + ln;
    float bv = bias[n];
#pragma unroll
    for (int m = 0; m < 2; ++m)
#pragma unroll
      for (int r = 0; r < 8; ++r) {
        float x = acc[m][t][r] + bv;
        acc[m][t][r] = x;
        sx[m][r]  += x;
        sq2[m][r] += x * x;
      }
  }
#pragma unroll
  for (int d = 8; d >= 1; d >>= 1)
#pragma unroll
    for (int m = 0; m < 2; ++m)
#pragma unroll
      for (int r = 0; r < 8; ++r) {
        sx[m][r]  += __shfl_xor(sx[m][r], d, 32);
        sq2[m][r] += __shfl_xor(sq2[m][r], d, 32);
      }
  if (ln == 0)
#pragma unroll
    for (int m = 0; m < 2; ++m)
#pragma unroll
      for (int r = 0; r < 8; ++r) {
        ssum[wave][m * 16 + hi * 8 + r] = sx[m][r];
        ssq[wave][m * 16 + hi * 8 + r]  = sq2[m][r];
      }
  __syncthreads();

  float mean[2][8], rinv[2][8];
#pragma unroll
  for (int m = 0; m < 2; ++m)
#pragma unroll
    for (int r = 0; r < 8; ++r) {
      int row = m * 16 + hi * 8 + r;
      float s1 = ssum[0][row], s2 = ssq[0][row];
#pragma unroll
      for (int w = 1; w < 8; ++w) { s1 += ssum[w][row]; s2 += ssq[w][row]; }
      float mu = s1 * (1.f / 1024.f);
      float var = s2 * (1.f / 1024.f) - mu * mu;
      mean[m][r] = mu;
      rinv[m][r] = rsqrtf(var + 1e-12f);
    }
#pragma unroll
  for (int t = 0; t < 8; ++t) {
    int n = wave * 128 + t * 16 + ln;
    float g = gamma[n], be = beta[n];
#pragma unroll
    for (int m = 0; m < 2; ++m)
#pragma unroll
      for (int r = 0; r < 8; ++r)
        out[((size_t)b * S_ + q0 + m * 16 + hi * 8 + r) * H_ + n] =
            (acc[m][t][r] - mean[m][r]) * rinv[m][r] * g + be;
  }
}

// ---------------------------------------------------------------------------
extern "C" void kernel_launch(void* const* d_in, const int* in_sizes, int n_in,
                              void* d_out, int out_size, void* d_ws, size_t ws_size,
                              hipStream_t stream) {
  const float* inputs = (const float*)d_in[0]; // [B,S,H]
  const float* masks  = (const float*)d_in[1]; // [B,S,S]
  const float* w      = (const float*)d_in[2]; // [2H,H]
  const float* bias   = (const float*)d_in[3]; // [H]
  const float* gamma  = (const float*)d_in[4]; // [H]
  const float* beta   = (const float*)d_in[5]; // [H]
  float* out = (float*)d_out;

  char* ws = (char*)d_ws;
  const size_t XR_BYTES = (size_t)B_ * S_ * H_ * 2; // 33554432
  __bf16* Xr  = (__bf16*)(ws);
  __bf16* Xt  = (__bf16*)(ws + XR_BYTES);
  __bf16* WbT = (__bf16*)(ws + 2 * XR_BYTES);
  __bf16* Ctx = (__bf16*)(ws + 2 * XR_BYTES + (size_t)2 * H_ * H_ * 2);

  const int N = B_ * S_ * H_;
  k_convert<<<dim3(N / 256), dim3(256), 0, stream>>>(inputs, Xr, N);

  k_transpose<<<dim3(H_ / 32, S_ / 32, B_), dim3(32, 8), 0, stream>>>(
      inputs, Xt, S_, H_, (size_t)S_ * H_, (size_t)S_ * H_);

  k_transpose<<<dim3(H_ / 32, (2 * H_) / 32, 1), dim3(32, 8), 0, stream>>>(
      w, WbT, 2 * H_, H_, 0, 0);

  k_attn<<<dim3(S_ / 32, B_), dim3(256), 0, stream>>>(Xr, Xt, masks, Ctx);

  k_match<<<dim3(S_ / 32, B_), dim3(256), 0, stream>>>(Xr, Ctx, WbT, bias, gamma,
                                                       beta, out);
}
